// FALR2_35794257445089
// MI455X (gfx1250) — matrix operand order repaired and checked by reading, not compiled, not run on hardware
//
#include <hip/hip_runtime.h>

// Problem constants (match reference): B=8, N=256, H=128, MID=128, OUT=128, ZD=256
#define Bc   8
#define Nc   256
#define Hc   128
#define MIDc 128
#define OUTc 128
#define ZDc  256

typedef __attribute__((ext_vector_type(16))) __bf16 v16bf;
typedef __attribute__((ext_vector_type(8)))  float  v8f;

union Frag16 {
    v16bf v;
    unsigned int u[8];
    uint4 q[2];
};

// f32 -> bf16 (round-half-up) pair pack: one v_add per value + one v_perm_b32.
__device__ __forceinline__ unsigned int pack_bf2(float a, float b) {
    unsigned int ua = __float_as_uint(a) + 0x8000u;
    unsigned int ub = __float_as_uint(b) + 0x8000u;
    // result bytes = [ua.b2, ua.b3, ub.b2, ub.b3]  (S0=ub -> sel 6,7 ; S1=ua -> sel 2,3)
    return __builtin_amdgcn_perm(ub, ua, 0x07060302u);
}
__device__ __forceinline__ unsigned short bf1(float a) {
    return (unsigned short)((__float_as_uint(a) + 0x8000u) >> 16);
}

// A-matrix (16 x K bf16 in LDS, row-major, pitch in bf16 elems) fragment.
// lane row = lane&15; hi = lane>>4. VGPR0..3: K=kbase+hi*8..+7 ; VGPR4..7: K=kbase+16+hi*8..+7
__device__ __forceinline__ void loadAfragLds(Frag16& f, const unsigned short* lds,
                                             int pitch, int row, int kbase, int hi) {
    const char* p = (const char*)(lds + (size_t)row * pitch + kbase + hi * 8);
    f.q[0] = *(const uint4*)(p);
    f.q[1] = *(const uint4*)(p + 32);
}

// B fragment from pre-swizzled bf16 fragment array in scratch: 2 coalesced b128 loads.
__device__ __forceinline__ void loadBfragWS(Frag16& f, const uint4* frag,
                                            int kc, int wave, int lane) {
    const uint4* p = frag + (((size_t)kc * 8 + wave) * 32 + lane) * 2;
    f.q[0] = p[0];
    f.q[1] = p[1];
}

// ---------------------------------------------------------------------------
// Prep: convert weight matrix W (K x 128, f32 row-major) into bf16 WMMA
// B-fragment layout: frag[((kc*8 + wt)*32 + lane)] = 8 packed u32.
// grid = (K/32)*8 blocks x 32 threads, runs once.
// ---------------------------------------------------------------------------
__global__ void FALR2_prepw_kernel(const float* __restrict__ W, uint4* __restrict__ frag) {
    int blk  = blockIdx.x;          // = kc*8 + wt
    int kc   = blk >> 3;
    int wt   = blk & 7;
    int lane = threadIdx.x;
    int n = lane & 15, hi = lane >> 4;
    int col = wt * 16 + n;
    unsigned int u[8];
#pragma unroll
    for (int v = 0; v < 8; ++v) {
        int kk = kc * 32 + (v >> 2) * 16 + hi * 8 + 2 * (v & 3);
        u[v] = pack_bf2(W[(size_t)kk * 128 + col], W[(size_t)(kk + 1) * 128 + col]);
    }
    uint4* dst = frag + ((size_t)blk * 32 + lane) * 2;
    uint4 q0, q1;
    q0.x = u[0]; q0.y = u[1]; q0.z = u[2]; q0.w = u[3];
    q1.x = u[4]; q1.y = u[5]; q1.z = u[6]; q1.w = u[7];
    dst[0] = q0;
    dst[1] = q1;
}

// ---------------------------------------------------------------------------
// msgg[b,m] = graph_fts[b,:] @ W_mg + b_mg   (8 x 128, trivial)
// ---------------------------------------------------------------------------
__global__ void FALR2_msgg_kernel(const float* __restrict__ graph,
                                  const float* __restrict__ Wmg,
                                  const float* __restrict__ bmg,
                                  float* __restrict__ msgg) {
    int b = blockIdx.x;
    int m = threadIdx.x;            // 128 threads
    float acc = bmg[m];
    const float* g = graph + (size_t)b * Hc;
#pragma unroll 4
    for (int k = 0; k < Hc; ++k) acc += g[k] * Wmg[(size_t)k * MIDc + m];
    msgg[(size_t)b * MIDc + m] = acc;
}

// ---------------------------------------------------------------------------
// Stage A: per (b, 16-row n-tile), three z-GEMMs via WMMA bf16:
//   base = z@W_m1 + b_m1 + msgg[b] + b_me   (all i/j-invariant terms folded)
//   m2   = z@W_m2 + b_m2
//   out1 = z@W_o1 + b_o1
// ---------------------------------------------------------------------------
__global__ __launch_bounds__(256) void FALR2_stageA_kernel(
    const float* __restrict__ node, const float* __restrict__ hidden,
    const uint4* __restrict__ fWm1, const float* __restrict__ bm1,
    const uint4* __restrict__ fWm2, const float* __restrict__ bm2,
    const uint4* __restrict__ fWo1, const float* __restrict__ bo1,
    const float* __restrict__ bme,  const float* __restrict__ msgg,
    float* __restrict__ baseWS, float* __restrict__ m2WS, float* __restrict__ o1WS)
{
    const int PZ = ZDc + 8;                                  // bf16 pitch, 16B-aligned rows
    __shared__ __align__(16) unsigned short zt[16 * PZ];     // z tile as bf16

    int t  = threadIdx.x;
    int b  = blockIdx.x >> 4;
    int n0 = (blockIdx.x & 15) * 16;

    // ---- cooperative stage of z = concat(node, hidden) rows into LDS (bf16) ----
    {
        int r   = t >> 4;            // tile row 0..15
        int c16 = (t & 15) * 16;     // column base, 16 floats per thread
        const float* src = (c16 < Hc)
            ? (node   + ((size_t)b * Nc + (n0 + r)) * Hc + c16)
            : (hidden + ((size_t)b * Nc + (n0 + r)) * Hc + (c16 - Hc));
        float4 f0 = ((const float4*)src)[0];
        float4 f1 = ((const float4*)src)[1];
        float4 f2 = ((const float4*)src)[2];
        float4 f3 = ((const float4*)src)[3];
        uint4 pA, pB;
        pA.x = pack_bf2(f0.x, f0.y); pA.y = pack_bf2(f0.z, f0.w);
        pA.z = pack_bf2(f1.x, f1.y); pA.w = pack_bf2(f1.z, f1.w);
        pB.x = pack_bf2(f2.x, f2.y); pB.y = pack_bf2(f2.z, f2.w);
        pB.z = pack_bf2(f3.x, f3.y); pB.w = pack_bf2(f3.z, f3.w);
        char* dst = (char*)zt + ((size_t)r * PZ + c16) * 2;
        ((uint4*)dst)[0] = pA;
        *(uint4*)(dst + 16) = pB;
    }
    __syncthreads();

    int lane = t & 31, wave = t >> 5;
    int nn = lane & 15, hi = lane >> 4;
    int col = wave * 16 + nn;

    v8f a0 = {}, a1 = {}, a2 = {};
#pragma unroll
    for (int kc = 0; kc < 8; ++kc) {
        Frag16 a, w;
        loadAfragLds(a, zt, PZ, nn, kc * 32, hi);
        loadBfragWS(w, fWm1, kc, wave, lane);
        a0 = __builtin_amdgcn_wmma_f32_16x16x32_bf16(false, a.v, false, w.v, (short)0, a0, false, false);
        loadBfragWS(w, fWm2, kc, wave, lane);
        a1 = __builtin_amdgcn_wmma_f32_16x16x32_bf16(false, a.v, false, w.v, (short)0, a1, false, false);
        loadBfragWS(w, fWo1, kc, wave, lane);
        a2 = __builtin_amdgcn_wmma_f32_16x16x32_bf16(false, a.v, false, w.v, (short)0, a2, false, false);
    }

    float cb = bm1[col] + msgg[(size_t)b * MIDc + col] + bme[col];
    float c2 = bm2[col];
    float co = bo1[col];
#pragma unroll
    for (int v = 0; v < 8; ++v) {
        size_t row = (size_t)b * Nc + n0 + v + 8 * hi;
        baseWS[row * MIDc + col] = a0[v] + cb;
        m2WS  [row * MIDc + col] = a1[v] + c2;
        o1WS  [row * OUTc + col] = a2[v] + co;
    }
}

// ---------------------------------------------------------------------------
// Stage B: per (b, 16-wide j-tile). Loop over all 256 source nodes i:
//   msgs[b,j,m] = max_i ( adj[b,i,j] ? edge[b,i,j,:]@W_me + m2[b,i,m] + base[b,j,m] : 0 )
// then out = out1 + msgs @ W_o2 + b_o2  (WMMA through LDS).
// Memory-bound: streams 268 MB of edge_fts exactly once (~11.5 us at 23.3 TB/s).
// ---------------------------------------------------------------------------
__global__ __launch_bounds__(256) void FALR2_stageB_kernel(
    const float* __restrict__ edge, const int* __restrict__ adj,
    const uint4* __restrict__ fWme, const uint4* __restrict__ fWo2,
    const float* __restrict__ bo2,
    const float* __restrict__ baseWS, const float* __restrict__ m2WS,
    const float* __restrict__ o1WS,
    float* __restrict__ out)
{
    const int PE = Hc + 8;                                      // bf16 pitch
    __shared__ __align__(16) unsigned short at[2][16 * PE];     // double-buffered edge tile
    __shared__ __align__(16) int adjrow[2][16];

    int t  = threadIdx.x;
    int b  = blockIdx.x >> 4;
    int j0 = (blockIdx.x & 15) * 16;

    int lane = t & 31, wave = t >> 5;
    int nn = lane & 15, hi = lane >> 4;
    int col = wave * 16 + nn;

    // W_me fragments resident in VGPRs for the whole i-loop (32 VGPRs)
    Frag16 wf[4];
#pragma unroll
    for (int kc = 0; kc < 4; ++kc) loadBfragWS(wf[kc], fWme, kc, wave, lane);

    float basev[8];
#pragma unroll
    for (int v = 0; v < 8; ++v)
        basev[v] = baseWS[((size_t)b * Nc + j0 + v + 8 * hi) * MIDc + col];

    float accm[8];
#pragma unroll
    for (int v = 0; v < 8; ++v) accm[v] = -__builtin_inff();

    int r  = t >> 4;           // cooperative-load row
    int c8 = (t & 15) * 8;     // 8 floats per thread

    for (int i = 0; i < Nc; ++i) {
        int p = i & 1;
        // ---- stage edge_fts[b, i, j0:j0+16, :] into LDS as bf16 ----
        const float* src = edge + (((size_t)b * Nc + i) * Nc + (j0 + r)) * Hc + c8;
        float4 f0 = ((const float4*)src)[0];
        float4 f1 = ((const float4*)src)[1];
        if (i + 1 < Nc) __builtin_prefetch(src + (size_t)Nc * Hc, 0, 3);  // next i slab
        uint4 pk;
        pk.x = pack_bf2(f0.x, f0.y); pk.y = pack_bf2(f0.z, f0.w);
        pk.z = pack_bf2(f1.x, f1.y); pk.w = pack_bf2(f1.z, f1.w);
        *(uint4*)((char*)at[p] + ((size_t)r * PE + c8) * 2) = pk;
        if (t < 16) adjrow[p][t] = adj[((size_t)b * Nc + i) * Nc + j0 + t];
        __syncthreads();   // single barrier per iteration (double-buffered)

        // ---- 16x16 msg_e tile: 4 chained K=32 bf16 WMMAs ----
        v8f d = {};
#pragma unroll
        for (int kc = 0; kc < 4; ++kc) {
            Frag16 a;
            loadAfragLds(a, at[p], PE, nn, kc * 32, hi);
            d = __builtin_amdgcn_wmma_f32_16x16x32_bf16(false, a.v, false, wf[kc].v, (short)0, d, false, false);
        }

        float v2 = m2WS[((size_t)b * Nc + i) * MIDc + col];
        int4 am0 = *(const int4*)&adjrow[p][hi * 8];
        int4 am1 = *(const int4*)&adjrow[p][hi * 8 + 4];
        int am[8] = { am0.x, am0.y, am0.z, am0.w, am1.x, am1.y, am1.z, am1.w };
#pragma unroll
        for (int v = 0; v < 8; ++v) {
            float val = d[v] + v2 + basev[v];
            accm[v] = fmaxf(accm[v], am[v] != 0 ? val : 0.0f);
        }
    }

    // ---- msgs tile -> LDS bf16 (reuse buffer 0), then msgs @ W_o2 via WMMA ----
#pragma unroll
    for (int v = 0; v < 8; ++v)
        at[0][(size_t)(v + 8 * hi) * PE + col] = bf1(accm[v]);
    __syncthreads();

    Frag16 w2[4];
#pragma unroll
    for (int kc = 0; kc < 4; ++kc) loadBfragWS(w2[kc], fWo2, kc, wave, lane);

    v8f d = {};
#pragma unroll
    for (int kc = 0; kc < 4; ++kc) {
        Frag16 a;
        loadAfragLds(a, at[0], PE, nn, kc * 32, hi);
        d = __builtin_amdgcn_wmma_f32_16x16x32_bf16(false, a.v, false, w2[kc].v, (short)0, d, false, false);
    }

    float bo = bo2[col];
#pragma unroll
    for (int v = 0; v < 8; ++v) {
        size_t idx = ((size_t)b * Nc + j0 + v + 8 * hi) * OUTc + col;
        out[idx] = o1WS[idx] + d[v] + bo;
    }
}

// ---------------------------------------------------------------------------
extern "C" void kernel_launch(void* const* d_in, const int* in_sizes, int n_in,
                              void* d_out, int out_size, void* d_ws, size_t ws_size,
                              hipStream_t stream) {
    (void)in_sizes; (void)n_in; (void)out_size; (void)ws_size;

    const float* node   = (const float*)d_in[0];
    const float* edge   = (const float*)d_in[1];
    const float* graph  = (const float*)d_in[2];
    const int*   adj    = (const int*)  d_in[3];
    const float* hidden = (const float*)d_in[4];
    const float* Wm1 = (const float*)d_in[5];  const float* bm1 = (const float*)d_in[6];
    const float* Wm2 = (const float*)d_in[7];  const float* bm2 = (const float*)d_in[8];
    const float* Wme = (const float*)d_in[9];  const float* bme = (const float*)d_in[10];
    const float* Wmg = (const float*)d_in[11]; const float* bmg = (const float*)d_in[12];
    const float* Wo1 = (const float*)d_in[13]; const float* bo1 = (const float*)d_in[14];
    const float* Wo2 = (const float*)d_in[15]; const float* bo2 = (const float*)d_in[16];
    float* outp = (float*)d_out;

    // scratch layout (floats):
    //  msgg 1024 | base 262144 | m2 262144 | out1 262144 |
    //  fWm1 16384 | fWm2 16384 | fWo1 16384 | fWme 8192 | fWo2 8192   (u32 counts)
    float* ws    = (float*)d_ws;
    float* msggp = ws;
    float* basep = msggp + (size_t)Bc * MIDc;
    float* m2p   = basep + (size_t)Bc * Nc * MIDc;
    float* o1p   = m2p   + (size_t)Bc * Nc * MIDc;
    unsigned int* fbase = (unsigned int*)(o1p + (size_t)Bc * Nc * OUTc);
    uint4* fWm1 = (uint4*)(fbase);
    uint4* fWm2 = (uint4*)(fbase + 16384);
    uint4* fWo1 = (uint4*)(fbase + 32768);
    uint4* fWme = (uint4*)(fbase + 49152);
    uint4* fWo2 = (uint4*)(fbase + 57344);

    // one-time weight swizzle to bf16 fragment layout
    FALR2_prepw_kernel<<<(ZDc / 32) * 8, 32, 0, stream>>>(Wm1, fWm1);
    FALR2_prepw_kernel<<<(ZDc / 32) * 8, 32, 0, stream>>>(Wm2, fWm2);
    FALR2_prepw_kernel<<<(ZDc / 32) * 8, 32, 0, stream>>>(Wo1, fWo1);
    FALR2_prepw_kernel<<<(Hc  / 32) * 8, 32, 0, stream>>>(Wme, fWme);
    FALR2_prepw_kernel<<<(MIDc/ 32) * 8, 32, 0, stream>>>(Wo2, fWo2);

    FALR2_msgg_kernel<<<Bc, MIDc, 0, stream>>>(graph, Wmg, bmg, msggp);
    FALR2_stageA_kernel<<<Bc * (Nc / 16), 256, 0, stream>>>(
        node, hidden, fWm1, bm1, fWm2, bm2, fWo1, bo1, bme, msggp, basep, m2p, o1p);
    FALR2_stageB_kernel<<<Bc * (Nc / 16), 256, 0, stream>>>(
        edge, adj, fWme, fWo2, bo2, basep, m2p, o1p, outp);
}
